// DiagonalSSM_6828998000786
// MI455X (gfx1250) — compile-verified, matrix operand order checked by reading, not run on hardware
//
#include <hip/hip_runtime.h>
#include <hip/hip_bf16.h>
#include <cmath>

// ---------------------------------------------------------------------------
// DiagonalSSM: y_t = tanh(a)*y_{t-1} + b*x_t  (scan over T), then y @ w_out^T + b_out
// B=8, T=4096, D=1024
// ---------------------------------------------------------------------------

#define B_ 8
#define T_ 4096
#define D_ 1024
#define NC 32            // chunks along T
#define CH (T_ / NC)     // 128 steps per chunk

typedef __attribute__((ext_vector_type(16))) __bf16 v16bf;
typedef __attribute__((ext_vector_type(8)))  __bf16 v8bf;
typedef __attribute__((ext_vector_type(8)))  float  v8f;

union BF16x16 { v16bf v; v8bf h[2]; };

// ---------------------------------------------------------------------------
// CDNA5 async global->LDS copy helpers (ASYNCcnt-tracked DMA path)
// ---------------------------------------------------------------------------
__device__ __forceinline__ unsigned lds_off(const void* p) {
  // generic LDS address = {aperture_hi, lds_offset[31:0]}
  return (unsigned)(uintptr_t)p;
}

__device__ __forceinline__ void async_copy64B(unsigned lds, const __bf16* g) {
  asm volatile(
      "global_load_async_to_lds_b128 %0, %1, off\n\t"
      "global_load_async_to_lds_b128 %0, %1, off offset:16\n\t"
      "global_load_async_to_lds_b128 %0, %1, off offset:32\n\t"
      "global_load_async_to_lds_b128 %0, %1, off offset:48"
      :: "v"(lds), "v"(g) : "memory");
}

__device__ __forceinline__ void async_copy32B(unsigned lds, const __bf16* g) {
  asm volatile(
      "global_load_async_to_lds_b128 %0, %1, off\n\t"
      "global_load_async_to_lds_b128 %0, %1, off offset:16"
      :: "v"(lds), "v"(g) : "memory");
}

__device__ __forceinline__ void wait_async0() {
#if __has_builtin(__builtin_amdgcn_s_wait_asynccnt)
  __builtin_amdgcn_s_wait_asynccnt(0);
#else
  asm volatile("s_wait_asynccnt 0x0" ::: "memory");
#endif
}

// ---------------------------------------------------------------------------
// Pass 1: per-chunk local scan, record chunk-final value F[b][c][d]
// ---------------------------------------------------------------------------
__global__ __launch_bounds__(256) void k_chunk_finals(
    const float* __restrict__ x, const float* __restrict__ a,
    const float* __restrict__ bvec, float* __restrict__ F) {
  const int tid = blockIdx.x * blockDim.x + threadIdx.x;   // B*NC*D threads
  const int d = tid & (D_ - 1);
  const int c = (tid >> 10) & (NC - 1);
  const int b = tid >> 15;
  const float ah = tanhf(a[d]);
  const float bb = bvec[d];
  const float* p = x + ((size_t)(b * T_ + c * CH)) * D_ + d;
  float acc = 0.0f;
#pragma unroll 8
  for (int t = 0; t < CH; ++t)
    acc = fmaf(ah, acc, bb * p[(size_t)t * D_]);
  F[tid] = acc;
}

// ---------------------------------------------------------------------------
// Pass 2: scan across chunk finals -> carry entering each chunk, C[b][c][d]
// ---------------------------------------------------------------------------
__global__ __launch_bounds__(256) void k_carries(
    const float* __restrict__ F, const float* __restrict__ a,
    float* __restrict__ C) {
  const int tid = blockIdx.x * blockDim.x + threadIdx.x;   // B*D threads
  const int d = tid & (D_ - 1);
  const int b = tid >> 10;
  float ah = tanhf(a[d]);
  float ap = ah;
#pragma unroll
  for (int i = 0; i < 7; ++i) ap = ap * ap;                // ah^128
  float carry = 0.0f;
#pragma unroll
  for (int c = 0; c < NC; ++c) {
    const int idx = (b * NC + c) * D_ + d;
    C[idx] = carry;
    carry = fmaf(ap, carry, F[idx]);
  }
}

// ---------------------------------------------------------------------------
// Pass 3: redo chunk scan with correct carry, emit y as bf16 [M=B*T][K=D]
// ---------------------------------------------------------------------------
__global__ __launch_bounds__(256) void k_scan_emit(
    const float* __restrict__ x, const float* __restrict__ a,
    const float* __restrict__ bvec, const float* __restrict__ C,
    __bf16* __restrict__ y) {
  const int tid = blockIdx.x * blockDim.x + threadIdx.x;   // B*NC*D threads
  const int d = tid & (D_ - 1);
  const int c = (tid >> 10) & (NC - 1);
  const int b = tid >> 15;
  const float ah = tanhf(a[d]);
  const float bb = bvec[d];
  const size_t row0 = (size_t)(b * T_ + c * CH);
  const float* p = x + row0 * D_ + d;
  __bf16* q = y + row0 * D_ + d;
  float acc = C[tid];
#pragma unroll 8
  for (int t = 0; t < CH; ++t) {
    acc = fmaf(ah, acc, bb * p[(size_t)t * D_]);
    q[(size_t)t * D_] = (__bf16)acc;
  }
}

// ---------------------------------------------------------------------------
// Convert w_out (fp32 [N][K]) -> bf16 [N][K]
// ---------------------------------------------------------------------------
__global__ __launch_bounds__(256) void k_cvt_w(
    const float* __restrict__ w, __bf16* __restrict__ wb, int n) {
  const int i = blockIdx.x * blockDim.x + threadIdx.x;
  if (i < n) wb[i] = (__bf16)w[i];
}

// ---------------------------------------------------------------------------
// GEMM: out[m,n] = sum_k y[m,k] * w[n,k] + bias[n]
// Block: 256 threads (8 waves, 4x2), 256x128 block tile, BK=32.
// Wave tile 64x64 = 4x4 WMMA accums. Double-buffered LDS fed by
// global_load_async_to_lds_b128 (ASYNCcnt), software-pipelined one tile ahead.
// ---------------------------------------------------------------------------
#define BM 256
#define BN 128
#define BK 32
#define LDP 40   // padded LDS row length (bf16 elems): conflict-free frag reads

__global__ __launch_bounds__(256) void k_gemm_bf16(
    const __bf16* __restrict__ A,   // [M][K] bf16 (y)
    const __bf16* __restrict__ Bw,  // [N][K] bf16 (w_out)
    const float* __restrict__ bias, // [N]
    float* __restrict__ Cout,       // [M][N]
    int M, int N, int K) {
  __shared__ __bf16 As[2][BM * LDP];
  __shared__ __bf16 Bs[2][BN * LDP];

  const int tid  = threadIdx.x;
  const int lane = tid & 31;
  const int wave = tid >> 5;
  const int wm   = wave >> 1;      // 0..3 (M direction, 64 rows each)
  const int wn   = wave & 1;       // 0..1 (N direction, 64 cols each)
  const int m0   = blockIdx.y * BM;
  const int n0   = blockIdx.x * BN;

  const int lrow = lane & 15;
  const int ksel = lane >> 4;

  // cooperative copy mapping
  //  A: 256 rows x 32 cols -> each thread copies one row's 64B
  //  B: 128 rows x 32 cols -> pairs of threads split a row, 32B each
  const int arow  = tid;           // 0..255
  const int brow  = tid >> 1;      // 0..127
  const int bhalf = tid & 1;       // 0..1 (16 bf16)

  const __bf16* gA = A  + (size_t)(m0 + arow) * K;
  const __bf16* gB = Bw + (size_t)(n0 + brow) * K + bhalf * 16;
  unsigned ldsA[2], ldsB[2];
#pragma unroll
  for (int s = 0; s < 2; ++s) {
    ldsA[s] = lds_off(&As[s][arow * LDP]);
    ldsB[s] = lds_off(&Bs[s][brow * LDP + bhalf * 16]);
  }

  v8f acc[4][4];
#pragma unroll
  for (int i = 0; i < 4; ++i)
#pragma unroll
    for (int j = 0; j < 4; ++j) acc[i][j] = (v8f){};

  const int nk = K / BK;

  // prologue: stage tile 0 into buffer 0
  async_copy64B(ldsA[0], gA);
  async_copy32B(ldsB[0], gB);
  wait_async0();
  __syncthreads();

  int buf = 0;
  for (int kt = 0; kt < nk; ++kt) {
    // prefetch next tile into the other buffer (DMA overlaps the WMMAs below)
    if (kt + 1 < nk) {
      const int kn = (kt + 1) * BK;
      async_copy64B(ldsA[buf ^ 1], gA + kn);
      async_copy32B(ldsB[buf ^ 1], gB + kn);
    }

    // fragment loads from current buffer
    BF16x16 af[4], bfr[4];
#pragma unroll
    for (int i = 0; i < 4; ++i) {
      const v8bf* p =
          (const v8bf*)(&As[buf][(wm * 64 + i * 16 + lrow) * LDP + ksel * 16]);
      af[i].h[0] = p[0]; af[i].h[1] = p[1];
    }
#pragma unroll
    for (int j = 0; j < 4; ++j) {
      const v8bf* p =
          (const v8bf*)(&Bs[buf][(wn * 64 + j * 16 + lrow) * LDP + ksel * 16]);
      bfr[j].h[0] = p[0]; bfr[j].h[1] = p[1];
    }

#pragma unroll
    for (int i = 0; i < 4; ++i)
#pragma unroll
      for (int j = 0; j < 4; ++j)
        acc[i][j] = __builtin_amdgcn_wmma_f32_16x16x32_bf16(
            false, af[i].v, false, bfr[j].v, (short)0, acc[i][j], false, false);

    wait_async0();       // next tile's DMA complete (own wave's issues)
    __syncthreads();     // all waves done reading current buffer + DMA visible
    buf ^= 1;
  }

  // epilogue: add bias, store fp32
#pragma unroll
  for (int j = 0; j < 4; ++j) {
    const int col = n0 + wn * 64 + j * 16 + lrow;
    const float bv = bias[col];
#pragma unroll
    for (int i = 0; i < 4; ++i) {
      const int rowbase = m0 + wm * 64 + i * 16 + ksel * 8;
#pragma unroll
      for (int r = 0; r < 8; ++r)
        Cout[(size_t)(rowbase + r) * N + col] = acc[i][j][r] + bv;
    }
  }
}

// ---------------------------------------------------------------------------
extern "C" void kernel_launch(void* const* d_in, const int* in_sizes, int n_in,
                              void* d_out, int out_size, void* d_ws, size_t ws_size,
                              hipStream_t stream) {
  const float* x    = (const float*)d_in[0];
  const float* a    = (const float*)d_in[1];
  const float* bvec = (const float*)d_in[2];
  const float* w    = (const float*)d_in[3];
  const float* bo   = (const float*)d_in[4];
  float* out = (float*)d_out;

  const size_t Y_BYTES = (size_t)B_ * T_ * D_ * sizeof(__bf16);  // 64 MB
  const size_t W_BYTES = (size_t)D_ * D_ * sizeof(__bf16);       // 2 MB
  const size_t F_BYTES = (size_t)B_ * NC * D_ * sizeof(float);   // 1 MB

  char* ws = (char*)d_ws;
  __bf16* y   = (__bf16*)(ws);
  __bf16* wb  = (__bf16*)(ws + Y_BYTES);
  float*  F   = (float*)(ws + Y_BYTES + W_BYTES);
  float*  C   = (float*)(ws + Y_BYTES + W_BYTES + F_BYTES);

  // w_out fp32 -> bf16
  k_cvt_w<<<(D_ * D_ + 255) / 256, 256, 0, stream>>>(w, wb, D_ * D_);

  // 3-pass chunked scan
  k_chunk_finals<<<(B_ * NC * D_) / 256, 256, 0, stream>>>(x, a, bvec, F);
  k_carries<<<(B_ * D_) / 256, 256, 0, stream>>>(F, a, C);
  k_scan_emit<<<(B_ * NC * D_) / 256, 256, 0, stream>>>(x, a, bvec, C, y);

  // GEMM: [32768 x 1024] = y[32768 x 1024] * w[1024 x 1024]^T + bias
  const int M = B_ * T_, N = D_, K = D_;
  dim3 grid(N / BN, M / BM);
  k_gemm_bf16<<<grid, 256, 0, stream>>>(y, wb, bo, out, M, N, K);
}
